// FactorNN_16561393893933
// MI455X (gfx1250) — matrix operand order repaired and checked by reading, not compiled, not run on hardware
//
#include <hip/hip_runtime.h>
#include <utility>

typedef __attribute__((ext_vector_type(16))) _Float16 v16h;
typedef __attribute__((ext_vector_type(8)))  float    v8f;

// ---------------------------------------------------------------------------
// GEMM: Y[M,COUT] (+)= X[M,CIN] @ W[CIN,COUT] + bias   (batched over blockIdx.y)
// One 16x16 WMMA tile per wave per column-tile; A fragment reused across tiles.
// W is staged transposed in LDS so each B fragment is one contiguous 32B read.
// B fragments are batched (groups of 4) into distinct registers so the ds_load
// clauses overlap and the 4 WMMAs issue back-to-back without full dscnt drains.
// ---------------------------------------------------------------------------
template<int CIN, int COUT, bool ACCUM>
__global__ __launch_bounds__(256)
void gemm_wmma(const float* __restrict__ X,
               const float* __restrict__ W,
               const float* __restrict__ bias,
               float* __restrict__ Y, int M)
{
    constexpr int NT = COUT / 16;
    const float* Wb = W + (size_t)blockIdx.y * (CIN * COUT);
    float*       Yb = Y + (size_t)blockIdx.y * (size_t)M * COUT;

    __shared__ __align__(32) _Float16 lWt[CIN * COUT];   // transposed: [COUT][CIN]
    for (int i = threadIdx.x; i < CIN * COUT; i += 256) {
        int k = i / COUT, n = i % COUT;
        lWt[n * CIN + k] = (_Float16)Wb[i];
    }
    __syncthreads();

    const int wave = threadIdx.x >> 5;
    const int lane = threadIdx.x & 31;
    const int hsel = lane >> 4;       // half-wave select
    const int lrow = lane & 15;
    const int m0   = blockIdx.x * 128 + wave * 16;
    if (m0 >= M) return;              // wave-uniform exit (EXEC stays all-ones)

    int row  = m0 + lrow;
    int rowc = row < M ? row : M - 1; // clamp for safe loads
    const float* xrow = X + (size_t)rowc * CIN;

    v8f acc[NT];
#pragma unroll
    for (int nt = 0; nt < NT; ++nt)
        acc[nt] = (v8f){0.f,0.f,0.f,0.f,0.f,0.f,0.f,0.f};

#pragma unroll
    for (int k0 = 0; k0 < CIN; k0 += 32) {
        // A fragment: contiguous K order (lanes 0-15: K=base..base+15, lanes 16-31: K=base+16..base+31)
        const float4* xp = (const float4*)(xrow + k0 + hsel * 16);
        float4 q0 = xp[0], q1 = xp[1], q2 = xp[2], q3 = xp[3];
        v16h a;
        a[0]=(_Float16)q0.x; a[1]=(_Float16)q0.y; a[2]=(_Float16)q0.z; a[3]=(_Float16)q0.w;
        a[4]=(_Float16)q1.x; a[5]=(_Float16)q1.y; a[6]=(_Float16)q1.z; a[7]=(_Float16)q1.w;
        a[8]=(_Float16)q2.x; a[9]=(_Float16)q2.y; a[10]=(_Float16)q2.z; a[11]=(_Float16)q2.w;
        a[12]=(_Float16)q3.x; a[13]=(_Float16)q3.y; a[14]=(_Float16)q3.z; a[15]=(_Float16)q3.w;

        const int kbB = k0 + hsel * 16;
#pragma unroll
        for (int ntb = 0; ntb < NT; ntb += 4) {
            v16h bf[4];
#pragma unroll
            for (int u = 0; u < 4; ++u) {
                int col = (ntb + u) * 16 + lrow;
                bf[u] = *(const v16h*)&lWt[col * CIN + kbB];  // 32B aligned
            }
#pragma unroll
            for (int u = 0; u < 4; ++u)
                acc[ntb + u] = __builtin_amdgcn_wmma_f32_16x16x32_f16(
                    false, a, false, bf[u], (short)0, acc[ntb + u], false, false);
        }
    }

    // C/D layout: lanes 0-15 -> N=lane, VGPR i -> M=i ; lanes 16-31 -> M=8+i
#pragma unroll
    for (int nt = 0; nt < NT; ++nt) {
        int n = nt * 16 + lrow;
        float bv = bias ? bias[n] : 0.f;
#pragma unroll
        for (int i = 0; i < 8; ++i) {
            int r = m0 + hsel * 8 + i;
            if (r < M) {
                size_t o = (size_t)r * COUT + n;
                float v = acc[nt][i] + bv;
                if (ACCUM) v += Yb[o];
                Yb[o] = v;
            }
        }
    }
}

// ---------------------------------------------------------------------------
// InstanceNorm (over row axis, per channel) -- deterministic two-pass + apply
// ---------------------------------------------------------------------------
template<int C>
__global__ void inorm_partial(const float* __restrict__ X, float* __restrict__ part, int M)
{
    __shared__ float ssum[256], ssq[256];
    const int R = 256 / C;
    int tid = threadIdx.x, c = tid % C, r0 = tid / C;
    float s = 0.f, q = 0.f;
    for (long r = (long)blockIdx.x * R + r0; r < M; r += (long)gridDim.x * R) {
        float v = X[r * C + c];
        s += v; q += v * v;
    }
    ssum[tid] = s; ssq[tid] = q;
    __syncthreads();
    if (tid < C) {
        float ts = 0.f, tq = 0.f;
        for (int i = tid; i < 256; i += C) { ts += ssum[i]; tq += ssq[i]; }
        part[blockIdx.x * 2 * C + c]     = ts;
        part[blockIdx.x * 2 * C + C + c] = tq;
    }
}

template<int C>
__global__ void inorm_finalize(const float* __restrict__ part, int nblk, int M,
                               float* __restrict__ meanrstd)
{
    int c = threadIdx.x;
    float s = 0.f, q = 0.f;
    for (int b = 0; b < nblk; ++b) {
        s += part[b * 2 * C + c];
        q += part[b * 2 * C + C + c];
    }
    float inv = 1.f / (float)M;
    float mean = s * inv;
    float var  = q * inv - mean * mean;      // biased var == jnp.var
    meanrstd[c]     = mean;
    meanrstd[C + c] = rsqrtf(var + 1e-5f);
}

template<int C>
__global__ void inorm_apply_relu(const float* __restrict__ X,
                                 const float* __restrict__ mr,
                                 float* __restrict__ Y, long tot)
{
    long i = (long)blockIdx.x * 256 + threadIdx.x;
    if (i >= tot) return;
    int c = (int)(i % C);
    float v = (X[i] - mr[c]) * mr[C + c];
    Y[i] = v > 0.f ? v : 0.f;
}

// ---------------------------------------------------------------------------
// Gather+mean+ReLU: out[t,h] = relu( mean_k proj[et[t,k]][idx[t,k]][h] + b1[h] )
// proj table is L2-resident (<=128MB vs 192MB L2); 4 targets / block.
// ---------------------------------------------------------------------------
__global__ __launch_bounds__(256)
void gather_mean_relu(const float* __restrict__ proj, const int* __restrict__ idx,
                      const int* __restrict__ et, const float* __restrict__ b1,
                      float* __restrict__ out, int T, int S)
{
    int h  = threadIdx.x & 63;
    int lt = threadIdx.x >> 6;
    long t = (long)blockIdx.x * 4 + lt;
    if (t >= T) return;

    int si[16], se[16];
    const int4* ip4 = (const int4*)(idx + t * 16);
    const int4* ep4 = (const int4*)(et  + t * 16);
#pragma unroll
    for (int k = 0; k < 4; ++k) { ((int4*)si)[k] = ip4[k]; ((int4*)se)[k] = ep4[k]; }

    float acc = 0.f;
#pragma unroll
    for (int k = 0; k < 16; ++k)
        acc += proj[((size_t)se[k] * S + (size_t)si[k]) * 64 + h];

    float v = acc * (1.f / 16.f) + b1[h];
    out[t * 64 + h] = v > 0.f ? v : 0.f;
}

// Final projection [N,128] @ [128,1] + b
__global__ void final_dot(const float* __restrict__ Yin, const float* __restrict__ w2,
                          const float* __restrict__ b2, float* __restrict__ out, int N)
{
    int r = blockIdx.x * 256 + threadIdx.x;
    if (r >= N) return;
    const float* yr = Yin + (size_t)r * 128;
    float acc = 0.f;
#pragma unroll 8
    for (int i = 0; i < 128; ++i) acc += yr[i] * w2[i];
    out[r] = acc + b2[0];
}

// ---------------------------------------------------------------------------
// Host orchestration
// ---------------------------------------------------------------------------
static void gemm(const float* X, const float* Wp, const float* bp, float* Y,
                 int M, int Cin, int Cout, int nb, bool accum, hipStream_t s)
{
    dim3 g((M + 127) / 128, nb);
    if (Cin == 128 && Cout == 64)
        gemm_wmma<128, 64, false><<<g, 256, 0, s>>>(X, Wp, bp, Y, M);
    else if (Cin == 64 && Cout == 64 && !accum)
        gemm_wmma<64, 64, false><<<g, 256, 0, s>>>(X, Wp, bp, Y, M);
    else if (Cin == 64 && Cout == 64 && accum)
        gemm_wmma<64, 64, true><<<g, 256, 0, s>>>(X, Wp, bp, Y, M);
    else
        gemm_wmma<64, 128, false><<<g, 256, 0, s>>>(X, Wp, bp, Y, M);
}

static void run_inorm_relu(const float* X, float* Y, int M, int C, float* red, hipStream_t s)
{
    const int NB = 240;
    float* mr = red + NB * 2 * 128;
    long tot = (long)M * C;
    int gap = (int)((tot + 255) / 256);
    if (C == 64) {
        inorm_partial<64><<<NB, 256, 0, s>>>(X, red, M);
        inorm_finalize<64><<<1, 64, 0, s>>>(red, NB, M, mr);
        inorm_apply_relu<64><<<gap, 256, 0, s>>>(X, mr, Y, tot);
    } else {
        inorm_partial<128><<<NB, 256, 0, s>>>(X, red, M);
        inorm_finalize<128><<<1, 128, 0, s>>>(red, NB, M, mr);
        inorm_apply_relu<128><<<gap, 256, 0, s>>>(X, mr, Y, tot);
    }
}

extern "C" void kernel_launch(void* const* d_in, const int* in_sizes, int n_in,
                              void* d_out, int out_size, void* d_ws, size_t ws_size,
                              hipStream_t stream)
{
    const float* nodef = (const float*)d_in[0];
    const float* hopf[2] = { (const float*)d_in[1], (const float*)d_in[2] };
    const int* idx_f2v[2] = { (const int*)d_in[3], (const int*)d_in[4] };
    const int* idx_v2f[2] = { (const int*)d_in[5], (const int*)d_in[6] };
    const int* et_f2v[2]  = { (const int*)d_in[7], (const int*)d_in[8] };
    const int* et_v2f[2]  = { (const int*)d_in[9], (const int*)d_in[10] };
    const float* P = (const float*)d_in[11];

    const int N = in_sizes[0] / 128;    // 100000
    const int F = in_sizes[1] / 64;     // 50000
    (void)n_in; (void)out_size; (void)ws_size;

    // ---- flattened param offsets (JAX pytree: sorted dict keys) ----
    const long P_FINAL_W1 = 0, P_FINAL_W2 = 8192, P_FINAL_B1 = 8320, P_FINAL_B2 = 8448;
    const long P_FMAP = 8449, FMAP_STRIDE = 4160;             // per type: W(4096)+b(64)
    const long P_LAYER0 = 16769, L_STRIDE = 111296;
    const long L_F2F = 0, F2F_STRIDE = 4160;
    const long L_F2V = 8320, MP_STRIDE = 24704;               // W1(20480)+W2(4096)+b1(64)+b2(64)
    const long L_V2F = 57728;
    const long L_V2V = 107136;
    const long P_NODE_W = 239361, P_NODE_B = 247553;

    // ---- workspace carve-up ----
    float* ws = (float*)d_ws;
    size_t o = 0;
    auto take = [&](size_t n) { float* p = ws + o; o += n; return p; };
    float* nodeA = take((size_t)N * 64);
    float* nodeB = take((size_t)N * 64);
    float* hopA[2] = { take((size_t)F * 64), take((size_t)F * 64) };
    float* hopB[2] = { take((size_t)F * 64), take((size_t)F * 64) };
    float* proj  = take((size_t)5 * N * 64);   // per-edge-type projection table
    float* msg   = take((size_t)N * 64);
    float* tmp   = take((size_t)N * 128);
    float* red   = take(240 * 2 * 128 + 256);

    auto iid = [&](const float* X, long wOff, long bOff, float* out,
                   int M, int Cin, int Cout) {
        gemm(X, P + wOff, P + bOff, tmp, M, Cin, Cout, 1, false, stream);
        run_inorm_relu(tmp, out, M, Cout, red, stream);
    };

    auto mp = [&](const float* src, int S, const int* idx, const int* et, int T,
                  long mpOff, float* dst) {
        const float* W1 = P + mpOff;             // [5][64][64]
        const float* W2 = P + mpOff + 20480;
        const float* b1 = P + mpOff + 20480 + 4096;
        const float* b2 = b1 + 64;
        gemm(src, W1, nullptr, proj, S, 64, 64, 5, false, stream);   // 5 batched GEMMs
        gather_mean_relu<<<dim3((T + 3) / 4), 256, 0, stream>>>(proj, idx, et, b1, msg, T, S);
        gemm(msg, W2, b2, dst, T, 64, 64, 1, true, stream);          // accumulate into dst
    };

    // ---- pipeline ----
    iid(nodef, P_NODE_W, P_NODE_B, nodeA, N, 128, 64);
    for (int j = 0; j < 2; ++j)
        iid(hopf[j], P_FMAP + j * FMAP_STRIDE, P_FMAP + j * FMAP_STRIDE + 4096,
            hopA[j], F, 64, 64);

    for (int l = 0; l < 2; ++l) {
        long base = P_LAYER0 + l * L_STRIDE;
        iid(nodeA, base + L_V2V, base + L_V2V + 4096, nodeB, N, 64, 64);
        for (int j = 0; j < 2; ++j)
            iid(hopA[j], base + L_F2F + j * F2F_STRIDE,
                base + L_F2F + j * F2F_STRIDE + 4096, hopB[j], F, 64, 64);
        for (int j = 0; j < 2; ++j) {
            mp(hopA[j], F, idx_f2v[j], et_f2v[j], N, base + L_F2V + j * MP_STRIDE, nodeB);
            mp(nodeA,   N, idx_v2f[j], et_v2f[j], F, base + L_V2F + j * MP_STRIDE, hopB[j]);
        }
        std::swap(nodeA, nodeB);
        std::swap(hopA[0], hopB[0]);
        std::swap(hopA[1], hopB[1]);
    }

    // final head: inorm_relu(nnode @ W1 + b1) @ W2 + b2
    gemm(nodeA, P + P_FINAL_W1, P + P_FINAL_B1, tmp, N, 64, 128, 1, false, stream);
    run_inorm_relu(tmp, tmp, N, 128, red, stream);   // in-place apply is safe
    final_dot<<<(N + 255) / 256, 256, 0, stream>>>(tmp, P + P_FINAL_W2, P + P_FINAL_B2,
                                                   (float*)d_out, N);
}
